// FineReviewDecoderAttention_14027363189385
// MI455X (gfx1250) — compile-verified, matrix operand order checked by reading, not run on hardware
//
#include <hip/hip_runtime.h>
#include <hip/hip_bf16.h>

typedef unsigned short u16;
typedef unsigned int   u32;
typedef __attribute__((ext_vector_type(16))) __bf16 v16bf;
typedef __attribute__((ext_vector_type(8)))  float  v8f;
typedef __attribute__((ext_vector_type(4)))  unsigned int v4u;
typedef __attribute__((ext_vector_type(8)))  int v8i;
typedef __attribute__((ext_vector_type(4)))  int v4i;
typedef int v4i_vs __attribute__((vector_size(16)));   // matches builtin param type

#define GLOBAL_AS __attribute__((address_space(1)))
#define LDS_AS    __attribute__((address_space(3)))

// ---- CDNA5 async / tensor data-mover availability (device pass only) ------
#if defined(__HIP_DEVICE_COMPILE__) && \
    __has_builtin(__builtin_amdgcn_global_load_async_to_lds_b128) && \
    __has_builtin(__builtin_amdgcn_s_wait_asynccnt)
#define HAVE_ASYNC_LDS 1
#endif

#if defined(__HIP_DEVICE_COMPILE__) && \
    __has_builtin(__builtin_amdgcn_tensor_load_to_lds) && \
    __has_builtin(__builtin_amdgcn_s_wait_tensorcnt)
#define HAVE_TDM 1
#if __has_include(<hip/amd_detail/amd_gfx1250_TDM.h>)
// amdgpu-toolchain (clang-23 / therock): 6-arg form with extra v8i group
#define TDM_LOAD_CALL(g0, g1, g2, g3)                                   \
    do {                                                                \
        v8i _z8 = {0, 0, 0, 0, 0, 0, 0, 0};                             \
        __builtin_amdgcn_tensor_load_to_lds((g0), (g1), (g2), (g3), _z8, 0); \
    } while (0)
#else
// ROCm 7.2 (clang-22): 5-arg form
#define TDM_LOAD_CALL(g0, g1, g2, g3)                                   \
    __builtin_amdgcn_tensor_load_to_lds((g0), (g1), (g2), (g3), 0)
#endif
#endif

#ifdef HAVE_ASYNC_LDS
// one 16-byte lane-chunk: global (AS1) -> LDS (AS3), ASYNCcnt-tracked
#define ASYNC_CP16(gsrc, ldst)                                          \
    __builtin_amdgcn_global_load_async_to_lds_b128(                     \
        (GLOBAL_AS v4i_vs*)(gsrc), (LDS_AS v4i_vs*)(ldst), 0, 0)
#endif

union Frag { v16bf v; u32 u[8]; };

__device__ __forceinline__ u16 f2bf(float x) {
    u32 u = __float_as_uint(x);
    return (u16)((u + 0x7FFFu + ((u >> 16) & 1u)) >> 16);   // round-to-nearest-even
}
__device__ __forceinline__ float bf2f(u16 h) {
    return __uint_as_float(((u32)h) << 16);
}

// packed f32x2 -> bf16x2 (single v_cvt_pk_bf16_f32 when available)
#if defined(__HIP_DEVICE_COMPILE__) && __has_builtin(__builtin_amdgcn_cvt_pk_bf16_f32)
#define HAVE_CVT_PK 1
typedef __attribute__((ext_vector_type(2))) __bf16 v2bf;
#endif
__device__ __forceinline__ u32 f2bf_pk(float lo, float hi) {
#ifdef HAVE_CVT_PK
    union { v2bf v; u32 u; } cv;
    cv.v = __builtin_amdgcn_cvt_pk_bf16_f32(lo, hi);
    return cv.u;
#else
    return (u32)f2bf(lo) | ((u32)f2bf(hi) << 16);
#endif
}

#ifdef HAVE_TDM
// Issue one TDM 2D tile load: tile_h rows x tile_w bf16 elements, global row
// stride row_stride_elems, into contiguous LDS at lds_ptr.  D# packing per
// ISA 8.3/8.4: g0 = {count=1 | lds_addr | global_addr | type=2},
// g1 = {data_size=1(2B), tensor_dim0/1, tile_dim0/1, tensor_dim0_stride}.
__device__ __forceinline__ void tdm_load_2d_bf16(const void* gaddr, void* lds_ptr,
                                                 u32 tile_w, u32 tile_h,
                                                 u32 row_stride_elems) {
    unsigned long long ga = (unsigned long long)gaddr;
    u32 lds_off = (u32)(unsigned long long)(LDS_AS char*)lds_ptr;
    v4u g0;
    g0.x = 1u;                                              // count=1 (user D#)
    g0.y = lds_off;                                         // LDS byte address
    g0.z = (u32)ga;                                         // global_addr[31:0]
    g0.w = ((u32)(ga >> 32) & 0x01FFFFFFu) | (2u << 30);    // addr[56:32] | type=2
    v8i g1;
    g1[0] = (int)(1u << 16);                                // data_size=1 -> 2 bytes
    g1[1] = (int)((tile_w & 0xFFFFu) << 16);                // tensor_dim0[15:0]
    g1[2] = (int)(((tile_w >> 16) & 0xFFFFu) |              // tensor_dim0[31:16]
                  ((tile_h & 0xFFFFu) << 16));              // tensor_dim1[15:0]
    g1[3] = (int)(((tile_h >> 16) & 0xFFFFu) |              // tensor_dim1[31:16]
                  ((tile_w & 0xFFFFu) << 16));              // tile_dim0
    g1[4] = (int)(tile_h & 0xFFFFu);                        // tile_dim1 (tile_dim2=0)
    g1[5] = (int)row_stride_elems;                          // tensor_dim0_stride[31:0]
    g1[6] = 0;                                              // stride hi / dim1_stride lo
    g1[7] = 0;
    v4i z4 = {0, 0, 0, 0};                                  // groups 2/3 unused (2D)
    TDM_LOAD_CALL(g0, g1, z4, z4);
}
#endif

// Gather one 16x32 bf16 WMMA fragment (A or B role) from an LDS tile stored
// [row][K] row-major. ISA 7.12.2: lane%16 = row, lane/16 selects K-halves,
// VGPR v holds packed K = (v>=4?16:0) + 8*(lane/16) + 2*(v&3) + {0,1}.
__device__ __forceinline__ v16bf load_frag(const u16* __restrict__ tile,
                                           int row, int ld, int k0) {
    Frag f;
    const int g = (threadIdx.x & 31) >> 4;
    const u16* p = tile + row * ld + k0;
#pragma unroll
    for (int v = 0; v < 8; ++v) {
        int kk = ((v & 4) << 2) + g * 8 + ((v & 3) << 1);
        f.u[v] = *(const u32*)(p + kk);
    }
    return f.v;
}

__device__ __forceinline__ v8f wmma_bf16(v16bf a, v16bf b, v8f c) {
    return __builtin_amdgcn_wmma_f32_16x16x32_bf16(
        /*neg_a=*/false, a, /*neg_b=*/false, b,
        /*c_mod=*/(short)0, c, /*reuse_a=*/false, /*reuse_b=*/false);
}

// ---------------------------------------------------------------------------
// Generic GEMM: Out[M,N] = X[M,K] @ W[N,K]^T + bias.  Block tile 128x64, BK=32.
// 256 threads = 8 waves; wave w owns rows [16w,16w+16), 4 n-tiles of 16.
// Grid: x = N-blocks (fast) so blocks sharing an A row-panel run adjacently
// and the big f32 activation matrix streams from HBM ~once (W stays in L2).
// ---------------------------------------------------------------------------
template <bool IN_BF16, bool OUT_BF16>
__global__ void __launch_bounds__(256)
gemm_wmma_kernel(const void* __restrict__ Xv, const float* __restrict__ W,
                 const float* __restrict__ bias, void* __restrict__ Outv,
                 int M, int N, int K) {
    __shared__ u16 As[128 * 32];
    __shared__ u16 Bs[64 * 32];

    const int tid  = threadIdx.x;
    const int wave = tid >> 5;
    const int lane = tid & 31;
    const int ln16 = lane & 15;
    const int g    = lane >> 4;
    const size_t bm = (size_t)blockIdx.y * 128;
    const size_t bn = (size_t)blockIdx.x * 64;

    v8f acc[4];
#pragma unroll
    for (int i = 0; i < 4; ++i)
#pragma unroll
        for (int j = 0; j < 8; ++j) acc[i][j] = 0.f;

    for (int k0 = 0; k0 < K; k0 += 32) {
        if (IN_BF16) {
            const u16* X = (const u16*)Xv;
#ifdef HAVE_ASYNC_LDS
#pragma unroll
            for (int i = 0; i < 2; ++i) {           // 512 x 16B chunks, ASYNCcnt
                int c   = tid + i * 256;
                int row = c >> 2;
                int col = (c & 3) * 8;
                ASYNC_CP16(X + (bm + row) * K + k0 + col, &As[row * 32 + col]);
            }
#else
#pragma unroll
            for (int i = 0; i < 8; ++i) {           // 2048 dword chunks
                int c   = tid + i * 256;
                int row = (c * 2) >> 5;
                int col = (c * 2) & 31;
                *(u32*)(&As[row * 32 + col]) =
                    *(const u32*)(X + (bm + row) * K + k0 + col);
            }
#endif
        } else {
            const float* X = (const float*)Xv;
#pragma unroll
            for (int i = 0; i < 4; ++i) {           // 1024 float4 chunks
                int c   = tid + i * 256;
                int row = c >> 3;
                int col = (c & 7) * 4;
                float4 f = *(const float4*)(X + (bm + row) * K + k0 + col);
                *(u32*)(&As[row * 32 + col])     = f2bf_pk(f.x, f.y);
                *(u32*)(&As[row * 32 + col + 2]) = f2bf_pk(f.z, f.w);
            }
        }
        // W is [N,K] row-major -> directly the [n][k] layout B-frags want.
#pragma unroll
        for (int i = 0; i < 2; ++i) {               // 512 float4 chunks
            int c   = tid + i * 256;
            int n   = c >> 3;
            int col = (c & 7) * 4;
            float4 f = *(const float4*)(W + (bn + n) * K + k0 + col);
            *(u32*)(&Bs[n * 32 + col])     = f2bf_pk(f.x, f.y);
            *(u32*)(&Bs[n * 32 + col + 2]) = f2bf_pk(f.z, f.w);
        }
        if (k0 + 32 < K) {                          // global_prefetch_b8 of next tiles
            __builtin_prefetch((const char*)Xv +
                ((bm + (tid >> 1)) * K + k0 + 32) * (IN_BF16 ? 2 : 4), 0, 1);
            __builtin_prefetch((const char*)(W + (bn + (tid >> 2)) * K + k0 + 32), 0, 1);
        }
#ifdef HAVE_ASYNC_LDS
        if (IN_BF16) __builtin_amdgcn_s_wait_asynccnt(0);
#endif
        __syncthreads();

        v16bf a = load_frag(As, wave * 16 + ln16, 32, 0);
#pragma unroll
        for (int nt = 0; nt < 4; ++nt) {
            v16bf b = load_frag(Bs, nt * 16 + ln16, 32, 0);
            acc[nt] = wmma_bf16(a, b, acc[nt]);
        }
        __syncthreads();
    }

#pragma unroll
    for (int nt = 0; nt < 4; ++nt) {
        size_t n  = bn + nt * 16 + ln16;
        float  bv = bias ? bias[n] : 0.f;
#pragma unroll
        for (int r = 0; r < 8; ++r) {
            size_t m   = bm + wave * 16 + r + g * 8;
            float  val = acc[nt][r] + bv;
            if (OUT_BF16) ((u16*)Outv)[m * N + n] = f2bf(val);
            else          ((float*)Outv)[m * N + n] = val;
        }
    }
}

// ---------------------------------------------------------------------------
// Token-level attention per (b,h,r): S = Q*K^T/8 -> mask -> softmax -> P*V.
// att stored bf16 as [B,H,Q,R,64].  K tile staged by the Tensor Data Mover,
// Q tile by async global->LDS copies, V transposed through VGPRs.
// ---------------------------------------------------------------------------
__global__ void __launch_bounds__(256)
attn_kernel(const u16* __restrict__ qb, const u16* __restrict__ kb,
            const u16* __restrict__ vb, const int* __restrict__ mask2,
            u16* __restrict__ att) {
    __shared__ u16 Qs[128 * 64];   // [q][d]
    __shared__ u16 Ks[64 * 64];    // [t][d]  (B-frag layout for S = Q*K^T)
    __shared__ u16 Vs[64 * 64];    // [d][t]  (transposed; B-frag layout for P*V)
    __shared__ u16 Ps[128 * 64];   // softmax probs

    const int tid  = threadIdx.x;
    const int wave = tid >> 5;
    const int lane = tid & 31;
    const int ln16 = lane & 15;
    const int g    = lane >> 4;

    const int blk = blockIdx.x;            // b*(16*20) + h*20 + r
    const int b   = blk / (16 * 20);
    const int rem = blk % (16 * 20);
    const int h   = rem / 20;
    const int r   = rem % 20;

    const u16* qbase = qb + (size_t)b * 128 * 1024 + h * 64;           // + q*1024 + d
    const u16* kbase = kb + (size_t)(b * 20 + r) * 64 * 1024 + h * 64; // + t*1024 + d
    const u16* vbase = vb + (size_t)(b * 20 + r) * 64 * 1024 + h * 64;

#ifdef HAVE_TDM
    if (wave == 0)                          // one TDM descriptor for the K tile
        tdm_load_2d_bf16(kbase, &Ks[0], /*tile_w=*/64, /*tile_h=*/64,
                         /*row_stride_elems=*/1024);
#else
#pragma unroll
    for (int i = 0; i < 8; ++i) {           // 2048 dword chunks of K
        int c = tid + i * 256;
        int t = (c * 2) >> 6, d = (c * 2) & 63;
        *(u32*)(&Ks[t * 64 + d]) = *(const u32*)(kbase + (size_t)t * 1024 + d);
    }
#endif

#ifdef HAVE_ASYNC_LDS
#pragma unroll
    for (int i = 0; i < 4; ++i) {           // 1024 x 16B chunks of Q, ASYNCcnt
        int c = tid + i * 256;
        int qi = c >> 3, d = (c & 7) * 8;
        ASYNC_CP16(qbase + (size_t)qi * 1024 + d, &Qs[qi * 64 + d]);
    }
#else
#pragma unroll
    for (int i = 0; i < 16; ++i) {          // 4096 dword chunks of Q
        int c = tid + i * 256;
        int qi = (c * 2) >> 6, d = (c * 2) & 63;
        *(u32*)(&Qs[qi * 64 + d]) = *(const u32*)(qbase + (size_t)qi * 1024 + d);
    }
#endif

#pragma unroll
    for (int i = 0; i < 16; ++i) {          // V transposed, element-wise
        int c = tid + i * 256;
        int t = c >> 6, dd = c & 63;
        Vs[dd * 64 + t] = vbase[(size_t)t * 1024 + dd];
    }

#ifdef HAVE_ASYNC_LDS
    __builtin_amdgcn_s_wait_asynccnt(0);
#endif
#ifdef HAVE_TDM
    if (wave == 0) __builtin_amdgcn_s_wait_tensorcnt(0);
#endif
    __syncthreads();

    v8f acc[4];
#pragma unroll
    for (int i = 0; i < 4; ++i)
#pragma unroll
        for (int j = 0; j < 8; ++j) acc[i][j] = 0.f;

#pragma unroll
    for (int ks = 0; ks < 2; ++ks) {
        v16bf a = load_frag(Qs, wave * 16 + ln16, 64, ks * 32);
#pragma unroll
        for (int nt = 0; nt < 4; ++nt) {
            v16bf bfr = load_frag(Ks, nt * 16 + ln16, 64, ks * 32);
            acc[nt] = wmma_bf16(a, bfr, acc[nt]);
        }
    }

    // scale + token padding mask (mask2==0 -> -1e20)
    const float inv_scale = 0.125f;         // 1/sqrt(64)
    const int* m2p = mask2 + (b * 20 + r) * 64;
#pragma unroll
    for (int nt = 0; nt < 4; ++nt) {
        bool pad = (m2p[nt * 16 + ln16] == 0);
#pragma unroll
        for (int rr = 0; rr < 8; ++rr)
            acc[nt][rr] = pad ? -1e20f : acc[nt][rr] * inv_scale;
    }

    // row softmax over T=64: row lives in one 16-lane half x 4 n-tiles
#pragma unroll
    for (int rr = 0; rr < 8; ++rr) {
        float mx = fmaxf(fmaxf(acc[0][rr], acc[1][rr]),
                         fmaxf(acc[2][rr], acc[3][rr]));
        for (int off = 1; off < 16; off <<= 1)
            mx = fmaxf(mx, __shfl_xor(mx, off, 16));
        float sum = 0.f;
#pragma unroll
        for (int nt = 0; nt < 4; ++nt) {
            float ex = __expf(acc[nt][rr] - mx);
            acc[nt][rr] = ex;
            sum += ex;
        }
        for (int off = 1; off < 16; off <<= 1)
            sum += __shfl_xor(sum, off, 16);
        float inv = 1.f / sum;
        int m = wave * 16 + rr + g * 8;
#pragma unroll
        for (int nt = 0; nt < 4; ++nt)
            Ps[m * 64 + nt * 16 + ln16] = f2bf(acc[nt][rr] * inv);
    }
    __syncthreads();

    v8f acc2[4];
#pragma unroll
    for (int i = 0; i < 4; ++i)
#pragma unroll
        for (int j = 0; j < 8; ++j) acc2[i][j] = 0.f;

#pragma unroll
    for (int ks = 0; ks < 2; ++ks) {
        v16bf a = load_frag(Ps, wave * 16 + ln16, 64, ks * 32);
#pragma unroll
        for (int nt = 0; nt < 4; ++nt) {
            v16bf bfr = load_frag(Vs, nt * 16 + ln16, 64, ks * 32);
            acc2[nt] = wmma_bf16(a, bfr, acc2[nt]);
        }
    }

    // att[b][h][q][r][dd]
    u16* abase = att + (((size_t)(b * 16 + h) * 128) * 20 + r) * 64;
#pragma unroll
    for (int nt = 0; nt < 4; ++nt) {
        int dd = nt * 16 + ln16;
#pragma unroll
        for (int rr = 0; rr < 8; ++rr) {
            int q = wave * 16 + rr + g * 8;
            abase[(size_t)q * 20 * 64 + dd] = f2bf(acc2[nt][rr]);
        }
    }
}

// ---------------------------------------------------------------------------
// e[m] = sum_n tanh((att @ sa_a)[m][n]) * sa_b[n]  with m = (b,h,q,r) flat.
// One big skinny WMMA GEMM (M=BHQR, K=N=64) + fused tanh-dot epilogue.
// ---------------------------------------------------------------------------
__global__ void __launch_bounds__(256)
pool_e_kernel(const u16* __restrict__ att, const float* __restrict__ sa_a,
              const float* __restrict__ sa_b, float* __restrict__ e) {
    __shared__ u16 As[128 * 64];
    __shared__ u16 Bs[64 * 64];     // [n][k] = sa_a[k][n]
    __shared__ float sb[64];

    const int tid  = threadIdx.x;
    const int wave = tid >> 5;
    const int lane = tid & 31;
    const int ln16 = lane & 15;
    const int g    = lane >> 4;
    const size_t bm = (size_t)blockIdx.x * 128;

#ifdef HAVE_ASYNC_LDS
#pragma unroll
    for (int i = 0; i < 4; ++i) {               // 1024 x 16B chunks of att
        int c = tid + i * 256;
        int row = c >> 3, col = (c & 7) * 8;
        ASYNC_CP16(att + (bm + row) * 64 + col, &As[row * 64 + col]);
    }
#else
#pragma unroll
    for (int i = 0; i < 16; ++i) {              // 4096 dword chunks of att
        int c = tid + i * 256;
        int row = (c * 2) >> 6, col = (c * 2) & 63;
        *(u32*)(&As[row * 64 + col]) =
            *(const u32*)(att + (bm + row) * 64 + col);
    }
#endif
    for (int c = tid; c < 4096; c += 256) {     // transpose sa_a into LDS
        int k = c >> 6, n = c & 63;
        Bs[n * 64 + k] = f2bf(sa_a[c]);
    }
    if (tid < 64) sb[tid] = sa_b[tid];
#ifdef HAVE_ASYNC_LDS
    __builtin_amdgcn_s_wait_asynccnt(0);
#endif
    __syncthreads();

    v8f acc[4];
#pragma unroll
    for (int i = 0; i < 4; ++i)
#pragma unroll
        for (int j = 0; j < 8; ++j) acc[i][j] = 0.f;

#pragma unroll
    for (int ks = 0; ks < 2; ++ks) {
        v16bf a = load_frag(As, wave * 16 + ln16, 64, ks * 32);
#pragma unroll
        for (int nt = 0; nt < 4; ++nt) {
            v16bf bfr = load_frag(Bs, nt * 16 + ln16, 64, ks * 32);
            acc[nt] = wmma_bf16(a, bfr, acc[nt]);
        }
    }

#pragma unroll
    for (int rr = 0; rr < 8; ++rr) {
        float part = 0.f;
#pragma unroll
        for (int nt = 0; nt < 4; ++nt)
            part += tanhf(acc[nt][rr]) * sb[nt * 16 + ln16];
        for (int off = 1; off < 16; off <<= 1)
            part += __shfl_xor(part, off, 16);
        if (ln16 == 0)
            e[bm + wave * 16 + rr + g * 8] = part;
    }
}

// ---------------------------------------------------------------------------
// Review-level softmax over R=20 (with -1e30*mask*any_zero) + weighted pool
// + head merge into bf16 [B,Q,DIM] for the output projection.
// ---------------------------------------------------------------------------
__global__ void __launch_bounds__(64)
pool_merge_kernel(const float* __restrict__ e, const int* __restrict__ mask,
                  const u16* __restrict__ att, u16* __restrict__ merged) {
    const int blk = blockIdx.x;        // (b*16 + h)*128 + q
    const int d   = threadIdx.x;       // 0..63
    const int b   = blk >> 11;
    const int rem = blk & 2047;
    const int h   = rem >> 7;
    const int q   = rem & 127;

    const float* ep = e + (size_t)blk * 20;
    const int*   mp = mask + b * 20;

    bool anyz = false;
#pragma unroll
    for (int r = 0; r < 20; ++r) anyz |= (mp[r] == 0);

    float ev[20];
    float mx = -3.4e38f;
#pragma unroll
    for (int r = 0; r < 20; ++r) {
        float val = ep[r] + (anyz ? (-1e30f * (float)mp[r]) : 0.f);
        ev[r] = val;
        mx = fmaxf(mx, val);
    }
    float sum = 0.f;
#pragma unroll
    for (int r = 0; r < 20; ++r) { ev[r] = __expf(ev[r] - mx); sum += ev[r]; }
    float inv = 1.f / sum;

    const u16* ap = att + (size_t)blk * 20 * 64 + d;
    float pooled = 0.f;
#pragma unroll
    for (int r = 0; r < 20; ++r) pooled += ev[r] * inv * bf2f(ap[r * 64]);

    merged[((size_t)b * 128 + q) * 1024 + h * 64 + d] = f2bf(pooled);
}

// ---------------------------------------------------------------------------
extern "C" void kernel_launch(void* const* d_in, const int* in_sizes, int n_in,
                              void* d_out, int out_size, void* d_ws, size_t ws_size,
                              hipStream_t stream) {
    (void)in_sizes; (void)n_in; (void)out_size; (void)ws_size;
    const float* query = (const float*)d_in[0];   // [32,128,1024]
    const float* key   = (const float*)d_in[1];   // [32,20,64,1024]
    const float* value = (const float*)d_in[2];
    const int*   mask  = (const int*)d_in[3];     // [32,20]
    const int*   mask2 = (const int*)d_in[4];     // [32,20,64]
    const float* Wq = (const float*)d_in[5];  const float* bq = (const float*)d_in[6];
    const float* Wk = (const float*)d_in[7];  const float* bk = (const float*)d_in[8];
    const float* Wv = (const float*)d_in[9];  const float* bv = (const float*)d_in[10];
    const float* Wo = (const float*)d_in[11]; const float* bo = (const float*)d_in[12];
    const float* sa_a = (const float*)d_in[13];
    const float* sa_b = (const float*)d_in[14];
    float* out = (float*)d_out;

    char* ws = (char*)d_ws;
    size_t off = 0;
    auto carve = [&](size_t bytes) {
        void* p = ws + off;
        off += (bytes + 255) & ~(size_t)255;
        return p;
    };
    u16*   qb     = (u16*)carve((size_t)32 * 128 * 1024 * 2);
    u16*   kb     = (u16*)carve((size_t)32 * 20 * 64 * 1024 * 2);
    u16*   vb     = (u16*)carve((size_t)32 * 20 * 64 * 1024 * 2);
    u16*   attb   = (u16*)carve((size_t)32 * 16 * 128 * 20 * 64 * 2);
    float* ebuf   = (float*)carve((size_t)32 * 16 * 128 * 20 * 4);
    u16*   merged = (u16*)carve((size_t)32 * 128 * 1024 * 2);

    const dim3 blk256(256);

    // q/k/v projections (f32 in -> bf16 out); grid.x = N-blocks for A locality
    gemm_wmma_kernel<false, true><<<dim3(1024 / 64, 4096 / 128), blk256, 0, stream>>>(
        query, Wq, bq, qb, 4096, 1024, 1024);
    gemm_wmma_kernel<false, true><<<dim3(1024 / 64, 40960 / 128), blk256, 0, stream>>>(
        key, Wk, bk, kb, 40960, 1024, 1024);
    gemm_wmma_kernel<false, true><<<dim3(1024 / 64, 40960 / 128), blk256, 0, stream>>>(
        value, Wv, bv, vb, 40960, 1024, 1024);

    // token attention per (b,h,r)
    attn_kernel<<<32 * 16 * 20, blk256, 0, stream>>>(qb, kb, vb, mask2, attb);

    // e = tanh(att @ sa_a) @ sa_b as one skinny WMMA GEMM
    pool_e_kernel<<<(32 * 16 * 128 * 20) / 128, blk256, 0, stream>>>(attb, sa_a, sa_b, ebuf);

    // review softmax + pooling + head merge
    pool_merge_kernel<<<32 * 16 * 128, dim3(64), 0, stream>>>(ebuf, mask, attb, merged);

    // output projection (bf16 in -> f32 out)
    gemm_wmma_kernel<true, false><<<dim3(1024 / 64, 4096 / 128), blk256, 0, stream>>>(
        merged, Wo, bo, out, 4096, 1024, 1024);
}
